// HeteroAttnGNN_48670569398802
// MI455X (gfx1250) — matrix operand order repaired and checked by reading, not compiled, not run on hardware
//
#include <hip/hip_runtime.h>
#include <hip/hip_bf16.h>

// ---------------- model constants ----------------
enum : int {
  NF = 100000, NC = 4000, EE = 500000, BB = 256,
  D_FACT = 1536, D_COMP = 27,
  HIDDEN = 128, HEADS = 4, CPH = 32,
  NUM_LAYERS = 2, TIME_DIM = 8, EDGE_DIM = 10,
};

typedef __attribute__((ext_vector_type(16))) __bf16 v16bf;
typedef __attribute__((ext_vector_type(8)))  float  v8f;

union Frag16 { v16bf v; unsigned u[8]; unsigned short h[16]; };

__device__ __forceinline__ unsigned short f2bf(float f) {
  unsigned u = __float_as_uint(f);
  u += 0x7FFFu + ((u >> 16) & 1u);           // round-to-nearest-even
  return (unsigned short)(u >> 16);
}
__device__ __forceinline__ unsigned pack2bf(float a, float b) {
  return (unsigned)f2bf(a) | ((unsigned)f2bf(b) << 16);
}
__device__ __forceinline__ unsigned f2ord(float f) {  // order-preserving map
  unsigned u = __float_as_uint(f);
  return (u & 0x80000000u) ? ~u : (u | 0x80000000u);
}
__device__ __forceinline__ float ord2f(unsigned o) {
  unsigned u = (o & 0x80000000u) ? (o & 0x7FFFFFFFu) : ~o;
  return __uint_as_float(u);
}

// ---------------- pack B (f32 row-major [K,N]) into WMMA bf16 B-fragment layout ----------------
// Bp[((kt*Nt + nt)*32 + lane)*8 + j] = pack(B[k, n], B[k+1, n]),
//   k = kt*32 + (lane>=16 ? 16 : 0) + 2j,  n = nt*16 + (lane&15)
__global__ void pack_b_kernel(const float* __restrict__ B, unsigned* __restrict__ Bp,
                              int KA, int KP, int N) {
  int gid = blockIdx.x * blockDim.x + threadIdx.x;
  int Nt = N >> 4;
  int total = (KP >> 5) * Nt * 32 * 8;
  if (gid >= total) return;
  int j    = gid & 7;
  int lane = (gid >> 3) & 31;
  int nt   = (gid >> 8) % Nt;
  int kt   = (gid >> 8) / Nt;
  int k = kt * 32 + ((lane >> 4) << 4) + 2 * j;
  int n = nt * 16 + (lane & 15);
  float v0 = (k     < KA) ? B[(size_t)k       * N + n] : 0.f;
  float v1 = (k + 1 < KA) ? B[(size_t)(k + 1) * N + n] : 0.f;
  Bp[gid] = pack2bf(v0, v1);
}

// ---------------- WMMA GEMM: C[M,128] = opt_relu(A[M,KA] @ B + bias) ----------------
// 64 rows (4 M-tiles) per block, 8 waves = 8 N-tiles. K staged in 128-wide bf16
// chunks, double-buffered in LDS (one barrier per chunk): next chunk's global
// loads are issued before the current chunk's 16 WMMAs, stores to the other
// LDS buffer overlap compute. Per 32-K step: 1 B fragment (2 global b128)
// reused by 4 WMMAs; all 4 A fragments (8 ds b128) hoisted ahead of the WMMAs.
template<int KA, int KP, bool RELU>
__global__ __launch_bounds__(256) void gemm_wmma_kernel(
    const float* __restrict__ A, const unsigned* __restrict__ Bp,
    const float* __restrict__ bias, float* __restrict__ C, int M)
{
  constexpr int KC = (KP < 128) ? KP : 128;          // K chunk staged in LDS
  constexpr int NCHUNK = KP / KC;
  constexpr int NV = (64 * KC) / (4 * 256);          // float4's per thread per chunk
  __shared__ __align__(16) unsigned short sA[2][64 * KC];

  const int tid = threadIdx.x;
  const int rowBase = blockIdx.x * 64;

  // speculative prefetch of the next row-block (global_prefetch_b8)
  if (tid < 64 && rowBase + 64 + tid < M)
    __builtin_prefetch(A + (size_t)(rowBase + 64 + tid) * KA, 0, 1);

  const int lane  = tid & 31;
  const int nt    = tid >> 5;          // N-tile per wave (N = 128)
  const int khalf = lane >> 4;
  const int mrow  = lane & 15;
  const int n     = nt * 16 + (lane & 15);

  float4 reg[NV];
  // staging helpers: chunk -> registers -> LDS (bf16)
  auto load_chunk = [&](int kc) {
#pragma unroll
    for (int v = 0; v < NV; ++v) {
      int i   = tid + v * 256;               // index of float4 within chunk
      int row = i / (KC / 4);
      int kq  = (i - row * (KC / 4)) * 4;
      int r   = rowBase + row;
      float4 val = {0.f, 0.f, 0.f, 0.f};
      if constexpr (KA == KP && (KA & 3) == 0) {
        if (r < M) val = *(const float4*)(A + (size_t)r * KA + (kc + kq));
      } else {
        int k = kc + kq;
        if (r < M) {
          val.x = (k + 0 < KA) ? A[(size_t)r * KA + k + 0] : 0.f;
          val.y = (k + 1 < KA) ? A[(size_t)r * KA + k + 1] : 0.f;
          val.z = (k + 2 < KA) ? A[(size_t)r * KA + k + 2] : 0.f;
          val.w = (k + 3 < KA) ? A[(size_t)r * KA + k + 3] : 0.f;
        }
      }
      reg[v] = val;
    }
  };
  auto store_chunk = [&](int buf) {
#pragma unroll
    for (int v = 0; v < NV; ++v) {
      int i   = tid + v * 256;
      int row = i / (KC / 4);
      int kq  = (i - row * (KC / 4)) * 4;
      uint2 w;
      w.x = pack2bf(reg[v].x, reg[v].y);
      w.y = pack2bf(reg[v].z, reg[v].w);
      *(uint2*)&sA[buf][row * KC + kq] = w;
    }
  };

  v8f acc[4];
#pragma unroll
  for (int mt = 0; mt < 4; ++mt) acc[mt] = (v8f){0.f,0.f,0.f,0.f,0.f,0.f,0.f,0.f};

  const unsigned* bbase = Bp + ((size_t)nt * 32 + lane) * 8;

  load_chunk(0);
  store_chunk(0);
  __syncthreads();
  int buf = 0;

  for (int ci = 0; ci < NCHUNK; ++ci) {
    const int kc = ci * KC;
    if (ci + 1 < NCHUNK) load_chunk(kc + KC);   // issue next chunk's global loads

#pragma unroll
    for (int kt = 0; kt < KC / 32; ++kt) {
      // B fragment: contiguous 32B per lane, shared across 4 M-tiles
      Frag16 bfr;
      const uint4* bp = (const uint4*)(bbase + (size_t)((kc >> 5) + kt) * 2048);
      uint4 y0 = bp[0], y1 = bp[1];
      bfr.u[0] = y0.x; bfr.u[1] = y0.y; bfr.u[2] = y0.z; bfr.u[3] = y0.w;
      bfr.u[4] = y1.x; bfr.u[5] = y1.y; bfr.u[6] = y1.z; bfr.u[7] = y1.w;
      // hoist all 4 A fragments (8 ds_load_b128), then 4 back-to-back WMMAs
      Frag16 af[4];
#pragma unroll
      for (int mt = 0; mt < 4; ++mt) {
        const int rb = (mt * 16 + mrow) * KC + kt * 32 + khalf * 16;
        uint4 x0 = *(const uint4*)&sA[buf][rb];
        uint4 x1 = *(const uint4*)&sA[buf][rb + 8];
        af[mt].u[0] = x0.x; af[mt].u[1] = x0.y; af[mt].u[2] = x0.z; af[mt].u[3] = x0.w;
        af[mt].u[4] = x1.x; af[mt].u[5] = x1.y; af[mt].u[6] = x1.z; af[mt].u[7] = x1.w;
      }
#pragma unroll
      for (int mt = 0; mt < 4; ++mt)
        acc[mt] = __builtin_amdgcn_wmma_f32_16x16x32_bf16(
            false, af[mt].v, false, bfr.v, (short)0, acc[mt], false, false);
    }

    if (ci + 1 < NCHUNK) {
      store_chunk(buf ^ 1);      // overlaps: target buffer last read in the previous chunk, before the barrier
      buf ^= 1;
    }
    __syncthreads();             // single barrier per chunk
  }

  const float bv = bias ? bias[n] : 0.f;
#pragma unroll
  for (int mt = 0; mt < 4; ++mt) {
#pragma unroll
    for (int i = 0; i < 8; ++i) {
      int m = rowBase + mt * 16 + i + khalf * 8;  // C layout: VGPR i -> M=i (+8 hi lanes)
      if (m < M) {
        float v = acc[mt][i] + bv;
        if (RELU) v = fmaxf(v, 0.f);
        C[(size_t)m * 128 + n] = v;
      }
    }
  }
}

// ---------------- edge time features ----------------
__global__ void edge_feat_kernel(const float* __restrict__ ea, const float* __restrict__ lamp,
                                 float* __restrict__ ef, int E_) {
  int e = blockIdx.x * blockDim.x + threadIdx.x;
  if (e >= E_) return;
  float s   = fminf(fmaxf(ea[(size_t)e * 2 + 0], -1.f), 1.f);
  float dec = fminf(fmaxf(ea[(size_t)e * 2 + 1],  0.f), 1.f);
  float lam = fmaxf(lamp[0], 1e-6f);
  float dt  = -logf(fmaxf(dec, 1e-6f)) / lam;
  float t   = log1pf(fmaxf(dt, 0.f)) / 30.f;
  float* o = ef + (size_t)e * EDGE_DIM;
  o[0] = s; o[1] = dec;
#pragma unroll
  for (int k = 1; k < TIME_DIM; ++k) o[1 + k] = sinf(t * (float)k);
  o[EDGE_DIM - 1] = t;
}

// ---------------- fills ----------------
__global__ void fill_f32_kernel(float* p, float v, int n) {
  int i = blockIdx.x * blockDim.x + threadIdx.x;
  if (i < n) p[i] = v;
}
__global__ void fill_u32_kernel(unsigned* p, unsigned v, int n) {
  int i = blockIdx.x * blockDim.x + threadIdx.x;
  if (i < n) p[i] = v;
}

// ---------------- edge pass 1: logits + segment max ----------------
__global__ __launch_bounds__(256) void edge_logits_kernel(
    const float* __restrict__ ef, const int* __restrict__ src, const int* __restrict__ dst,
    const float* __restrict__ gl, const float* __restrict__ gr,
    const float* __restrict__ We, const float* __restrict__ be, const float* __restrict__ att,
    float* __restrict__ logits, unsigned* __restrict__ maxb, int E_)
{
  __shared__ float sWe[EDGE_DIM * 128];
  __shared__ float sbe[128];
  __shared__ float satt[128];
  for (int i = threadIdx.x; i < EDGE_DIM * 128; i += 256) sWe[i] = We[i];
  for (int i = threadIdx.x; i < 128; i += 256) { sbe[i] = be[i]; satt[i] = att[i]; }
  __syncthreads();

  int gid = blockIdx.x * 256 + threadIdx.x;
  if (gid >= E_ * HEADS) return;
  int e = gid >> 2, h = gid & 3;
  int s = src[e], d = dst[e];
  float efv[EDGE_DIM];
#pragma unroll
  for (int k = 0; k < EDGE_DIM; ++k) efv[k] = ef[(size_t)e * EDGE_DIM + k];
  const float* glp = gl + (size_t)s * 128 + h * 32;
  const float* grp = gr + (size_t)d * 128 + h * 32;
  float acc = 0.f;
#pragma unroll 4
  for (int c = 0; c < CPH; ++c) {
    int cc = h * 32 + c;
    float ge = sbe[cc];
#pragma unroll
    for (int k = 0; k < EDGE_DIM; ++k) ge = fmaf(efv[k], sWe[k * 128 + cc], ge);
    float x = glp[c] + grp[c] + ge;
    x = (x > 0.f) ? x : 0.2f * x;          // leaky_relu
    acc = fmaf(x, satt[cc], acc);
  }
  logits[gid] = acc;
  atomicMax(&maxb[d * HEADS + h], f2ord(acc));
}

// ---------------- edge pass 2: exp + segment sum ----------------
__global__ void edge_softmax_kernel(const int* __restrict__ dst, float* __restrict__ logits,
                                    const unsigned* __restrict__ maxb, float* __restrict__ denom, int E_)
{
  int gid = blockIdx.x * blockDim.x + threadIdx.x;
  if (gid >= E_ * HEADS) return;
  int e = gid >> 2, h = gid & 3;
  int d = dst[e];
  float m = ord2f(maxb[d * HEADS + h]);
  float z = expf(logits[gid] - m);
  logits[gid] = z;
  atomicAdd(&denom[d * HEADS + h], z);
}

// ---------------- edge pass 3: scatter messages ----------------
__global__ void edge_scatter_kernel(const int* __restrict__ src, const int* __restrict__ dst,
                                    const float* __restrict__ logits, const float* __restrict__ denom,
                                    const float* __restrict__ gl, float* __restrict__ outm, int E_)
{
  int gid = blockIdx.x * blockDim.x + threadIdx.x;
  if (gid >= E_ * HEADS) return;
  int e = gid >> 2, h = gid & 3;
  int s = src[e], d = dst[e];
  float alpha = logits[gid] / (denom[d * HEADS + h] + 1e-16f);
  const float* glp = gl + (size_t)s * 128 + h * 32;
  float* op = outm + (size_t)d * 128 + h * 32;
#pragma unroll
  for (int c = 0; c < CPH; ++c) atomicAdd(&op[c], alpha * glp[c]);
}

// ---------------- node update: residual + relu + layernorm ----------------
__global__ __launch_bounds__(128) void node_update_kernel(
    float* __restrict__ hcv, const float* __restrict__ outm,
    const float* __restrict__ bo, const float* __restrict__ g, const float* __restrict__ b)
{
  int n = blockIdx.x, c = threadIdx.x;
  float o = outm[(size_t)n * 128 + c] + bo[c];
  o = fmaxf(o, 0.f);
  float v = hcv[(size_t)n * 128 + c] + o;
  float s1 = v, s2 = v * v;
  int lane = c & 31, wid = c >> 5;
  for (int off = 16; off > 0; off >>= 1) {
    s1 += __shfl_down(s1, off, 32);
    s2 += __shfl_down(s2, off, 32);
  }
  __shared__ float r1[4], r2[4];
  __shared__ float mu, rstd;
  if (lane == 0) { r1[wid] = s1; r2[wid] = s2; }
  __syncthreads();
  if (c == 0) {
    float t1 = r1[0] + r1[1] + r1[2] + r1[3];
    float t2 = r2[0] + r2[1] + r2[2] + r2[3];
    float m = t1 / 128.f;
    mu = m;
    rstd = rsqrtf(t2 / 128.f - m * m + 1e-5f);
  }
  __syncthreads();
  hcv[(size_t)n * 128 + c] = (v - mu) * rstd * g[c] + b[c];
}

// ---------------- pooling ----------------
__global__ void pool_accum_kernel(const float* __restrict__ x, const int* __restrict__ batch,
                                  float* __restrict__ pool, float* __restrict__ cnt, int Nn)
{
  int n = blockIdx.x * blockDim.x + threadIdx.x;
  if (n >= Nn) return;
  int b = batch[n];
  const float* xp = x + (size_t)n * 128;
  float* pp = pool + (size_t)b * 128;
#pragma unroll 8
  for (int c = 0; c < 128; ++c) atomicAdd(&pp[c], xp[c]);
  atomicAdd(&cnt[b], 1.f);
}
__global__ void pool_div_kernel(float* __restrict__ pool, const float* __restrict__ cnt, int total) {
  int i = blockIdx.x * blockDim.x + threadIdx.x;
  if (i < total) pool[i] /= fmaxf(cnt[i >> 7], 1.f);
}

// ---------------- gated readout ----------------
__global__ __launch_bounds__(128) void gate_kernel(
    const float* __restrict__ pf, const float* __restrict__ pc,
    const float* __restrict__ Wg, const float* __restrict__ bg, float* __restrict__ hg)
{
  int b = blockIdx.x, t = threadIdx.x;
  float pfv = pf[(size_t)b * 128 + t], pcv = pc[(size_t)b * 128 + t];
  float partial = pfv * Wg[t] + pcv * Wg[128 + t];
  int lane = t & 31, wid = t >> 5;
  for (int off = 16; off > 0; off >>= 1) partial += __shfl_down(partial, off, 32);
  __shared__ float r[4];
  __shared__ float gs;
  if (lane == 0) r[wid] = partial;
  __syncthreads();
  if (t == 0) {
    float ssum = r[0] + r[1] + r[2] + r[3] + bg[0];
    gs = 1.f / (1.f + expf(-ssum));
  }
  __syncthreads();
  float g = gs;
  hg[(size_t)b * 128 + t] = g * pfv + (1.f - g) * pcv;
}

// ---------------- final head: out[b] = tmp[b,:] . W2 + b2 ----------------
__global__ __launch_bounds__(32) void final_kernel(
    const float* __restrict__ tmp, const float* __restrict__ W2,
    const float* __restrict__ b2, float* __restrict__ out)
{
  int b = blockIdx.x, lane = threadIdx.x;
  float s = 0.f;
  for (int i = lane; i < 128; i += 32) s += tmp[(size_t)b * 128 + i] * W2[i];
  for (int off = 16; off > 0; off >>= 1) s += __shfl_down(s, off, 32);
  if (lane == 0) out[b] = s + b2[0];
}

// ---------------- launcher ----------------
extern "C" void kernel_launch(void* const* d_in, const int* in_sizes, int n_in,
                              void* d_out, int out_size, void* d_ws, size_t ws_size,
                              hipStream_t stream)
{
  (void)in_sizes; (void)n_in; (void)out_size; (void)ws_size;
  const float* x_fact    = (const float*)d_in[0];
  const float* x_comp    = (const float*)d_in[1];
  const float* edge_attr = (const float*)d_in[2];
  const int*   eidx      = (const int*)d_in[3];
  const int*   batch_f   = (const int*)d_in[4];
  const int*   batch_c   = (const int*)d_in[5];
  const float* lam       = (const float*)d_in[6];
  const float* Wf = (const float*)d_in[7];  const float* bf = (const float*)d_in[8];
  const float* Wc = (const float*)d_in[9];  const float* bc = (const float*)d_in[10];
  const float* Wl = (const float*)d_in[11]; const float* bl = (const float*)d_in[12];
  const float* Wr = (const float*)d_in[13]; const float* br = (const float*)d_in[14];
  const float* We = (const float*)d_in[15]; const float* be = (const float*)d_in[16];
  const float* att = (const float*)d_in[17]; const float* bias_out = (const float*)d_in[18];
  const float* ln_g = (const float*)d_in[19]; const float* ln_b = (const float*)d_in[20];
  const float* Wg = (const float*)d_in[21]; const float* bg = (const float*)d_in[22];
  const float* W1 = (const float*)d_in[23]; const float* b1 = (const float*)d_in[24];
  const float* W2 = (const float*)d_in[25]; const float* b2 = (const float*)d_in[26];
  const int* src = eidx;
  const int* dst = eidx + EE;
  float* out = (float*)d_out;

  char* base = (char*)d_ws;
  size_t off = 0;
  auto alloc = [&](size_t bytes) -> void* {
    void* p = base + off;
    off = (off + bytes + 255) & ~(size_t)255;
    return p;
  };
  float*    ef     = (float*)   alloc((size_t)EE * EDGE_DIM * 4);
  float*    hfbuf  = (float*)   alloc((size_t)NF * 128 * 4);
  float*    hcbuf  = (float*)   alloc((size_t)NC * 128 * 4);
  float*    glbuf  = (float*)   alloc((size_t)NF * 128 * 4);
  float*    grbuf  = (float*)   alloc((size_t)NC * 128 * 4);
  float*    logits = (float*)   alloc((size_t)EE * HEADS * 4);
  unsigned* maxb   = (unsigned*)alloc((size_t)NC * HEADS * 4);
  float*    denom  = (float*)   alloc((size_t)NC * HEADS * 4);
  float*    outm   = (float*)   alloc((size_t)NC * 128 * 4);
  float*    pf     = (float*)   alloc((size_t)BB * 128 * 4);
  float*    pc     = (float*)   alloc((size_t)BB * 128 * 4);
  float*    cntf   = (float*)   alloc((size_t)BB * 4);
  float*    cntc   = (float*)   alloc((size_t)BB * 4);
  float*    hg     = (float*)   alloc((size_t)BB * 128 * 4);
  float*    tmp    = (float*)   alloc((size_t)BB * 128 * 4);
  unsigned* WfP    = (unsigned*)alloc((size_t)1536 * 128 * 2);
  unsigned* WcP    = (unsigned*)alloc((size_t)32 * 128 * 2);
  unsigned* WlP    = (unsigned*)alloc((size_t)2 * 128 * 128 * 2);
  unsigned* WrP    = (unsigned*)alloc((size_t)2 * 128 * 128 * 2);
  unsigned* W1P    = (unsigned*)alloc((size_t)128 * 128 * 2);

  // ---- pack weights into WMMA bf16 fragment layout ----
  {
    int tot = (1536 / 32) * 8 * 256;
    pack_b_kernel<<<(tot + 255) / 256, 256, 0, stream>>>(Wf, WfP, 1536, 1536, 128);
    tot = 1 * 8 * 256;
    pack_b_kernel<<<(tot + 255) / 256, 256, 0, stream>>>(Wc, WcP, 27, 32, 128);
    tot = (128 / 32) * 8 * 256;
    for (int l = 0; l < NUM_LAYERS; ++l) {
      pack_b_kernel<<<(tot + 255) / 256, 256, 0, stream>>>(Wl + (size_t)l * 16384, WlP + (size_t)l * 8192, 128, 128, 128);
      pack_b_kernel<<<(tot + 255) / 256, 256, 0, stream>>>(Wr + (size_t)l * 16384, WrP + (size_t)l * 8192, 128, 128, 128);
    }
    pack_b_kernel<<<(tot + 255) / 256, 256, 0, stream>>>(W1, W1P, 128, 128, 128);
  }

  // ---- edge features ----
  edge_feat_kernel<<<(EE + 255) / 256, 256, 0, stream>>>(edge_attr, lam, ef, EE);

  // ---- node encoders (WMMA) ----
  gemm_wmma_kernel<1536, 1536, true><<<(NF + 63) / 64, 256, 0, stream>>>(x_fact, WfP, bf, hfbuf, NF);
  gemm_wmma_kernel<27, 32, true><<<(NC + 63) / 64, 256, 0, stream>>>(x_comp, WcP, bc, hcbuf, NC);

  // ---- GATv2 layers ----
  int eg = (EE * HEADS + 255) / 256;
  for (int l = 0; l < NUM_LAYERS; ++l) {
    gemm_wmma_kernel<128, 128, false><<<(NF + 63) / 64, 256, 0, stream>>>(hfbuf, WlP + (size_t)l * 8192, bl + l * 128, glbuf, NF);
    gemm_wmma_kernel<128, 128, false><<<(NC + 63) / 64, 256, 0, stream>>>(hcbuf, WrP + (size_t)l * 8192, br + l * 128, grbuf, NC);
    fill_u32_kernel<<<(NC * HEADS + 255) / 256, 256, 0, stream>>>(maxb, 0u, NC * HEADS);
    fill_f32_kernel<<<(NC * HEADS + 255) / 256, 256, 0, stream>>>(denom, 0.f, NC * HEADS);
    fill_f32_kernel<<<(NC * 128 + 255) / 256, 256, 0, stream>>>(outm, 0.f, NC * 128);
    edge_logits_kernel<<<eg, 256, 0, stream>>>(ef, src, dst, glbuf, grbuf,
                                               We + (size_t)l * EDGE_DIM * 128, be + l * 128,
                                               att + l * 128, logits, maxb, EE);
    edge_softmax_kernel<<<eg, 256, 0, stream>>>(dst, logits, maxb, denom, EE);
    edge_scatter_kernel<<<eg, 256, 0, stream>>>(src, dst, logits, denom, glbuf, outm, EE);
    node_update_kernel<<<NC, 128, 0, stream>>>(hcbuf, outm, bias_out + l * 128, ln_g + l * 128, ln_b + l * 128);
  }

  // ---- gated readout ----
  fill_f32_kernel<<<(BB * 128 + 255) / 256, 256, 0, stream>>>(pf, 0.f, BB * 128);
  fill_f32_kernel<<<(BB * 128 + 255) / 256, 256, 0, stream>>>(pc, 0.f, BB * 128);
  fill_f32_kernel<<<1, 256, 0, stream>>>(cntf, 0.f, BB);
  fill_f32_kernel<<<1, 256, 0, stream>>>(cntc, 0.f, BB);
  pool_accum_kernel<<<(NF + 255) / 256, 256, 0, stream>>>(hfbuf, batch_f, pf, cntf, NF);
  pool_accum_kernel<<<(NC + 255) / 256, 256, 0, stream>>>(hcbuf, batch_c, pc, cntc, NC);
  pool_div_kernel<<<(BB * 128 + 255) / 256, 256, 0, stream>>>(pf, cntf, BB * 128);
  pool_div_kernel<<<(BB * 128 + 255) / 256, 256, 0, stream>>>(pc, cntc, BB * 128);
  gate_kernel<<<BB, 128, 0, stream>>>(pf, pc, Wg, bg, hg);
  gemm_wmma_kernel<128, 128, true><<<(BB + 63) / 64, 256, 0, stream>>>(hg, W1P, b1, tmp, BB);
  final_kernel<<<BB, 32, 0, stream>>>(tmp, W2, b2, out);
}
